// DefromConvV3_41704132444210
// MI455X (gfx1250) — matrix operand order repaired and checked by reading, not compiled, not run on hardware
//
#include <hip/hip_runtime.h>
#include <hip/hip_bf16.h>
#include <math.h>

// ---------------- shapes ----------------
#define BB   4
#define CC   256
#define GG   8
#define CG   32
#define HH   64
#define WW   64
#define HW   4096
#define MPAD 224      // 216 off+mask channels padded to 14x16
#define WPAD 66       // 64 + 2*pad

typedef __attribute__((ext_vector_type(16))) __bf16 v16bf;
typedef __attribute__((ext_vector_type(8)))  float  v8f;

#define DEV __device__ __forceinline__

// gfx1250 async global->LDS DMA (ASYNCcnt), guarded so compile never breaks.
// Builtin signature (from hipcc diagnostic): arg0 = global ptr to int __vector(16B),
// arg1 = LDS ptr to same vector type, then two imm ints (offset, cpol).
#if defined(__gfx1250__) && __has_builtin(__builtin_amdgcn_global_load_async_to_lds_b128)
#define HAVE_ASYNC_LDS 1
typedef int v4i_vs __attribute__((vector_size(16)));
#define AS1V(p) ((__attribute__((address_space(1))) v4i_vs*)(p))
#define AS3V(p) ((__attribute__((address_space(3))) v4i_vs*)(p))
#else
#define HAVE_ASYNC_LDS 0
#endif

static DEV void async_wait_all() {
#if HAVE_ASYNC_LDS
#if __has_builtin(__builtin_amdgcn_s_wait_asynccnt)
  __builtin_amdgcn_s_wait_asynccnt(0);
#else
  asm volatile("s_wait_asynccnt 0x0" ::: "memory");
#endif
#endif
}

static DEV __bf16 f2bf(float f) {
  unsigned int u = __builtin_bit_cast(unsigned int, f);
  unsigned int r = u + 0x7FFFu + ((u >> 16) & 1u);   // round-to-nearest-even
  unsigned short h = (unsigned short)(r >> 16);
  return __builtin_bit_cast(__bf16, h);
}

// A-matrix 16x32 bf16 fragment per ISA layout:
// lane<16: row M=lane,    K = {0..7, 16..23}
// lane>=16: row M=lane-16, K = {8..15, 24..31}
static DEV v16bf load_a_frag(const __bf16* As, int lda, int row0, int lane) {
  int r  = row0 + (lane & 15);
  int kb = (lane < 16) ? 0 : 8;
  const __bf16* p = As + r * lda + kb;
  v16bf a;
#pragma unroll
  for (int e = 0; e < 8; ++e) { a[e] = p[e]; a[8 + e] = p[16 + e]; }
  return a;
}

// B-matrix 32x16 bf16 fragment; LDS tile stored transposed Bs[n][k]:
// lane<16: col N=lane, K={0..7,16..23} ; lane>=16: col N=lane-16, K={8..15,24..31}
static DEV v16bf load_b_frag(const __bf16* BsT, int ldb, int col0, int lane) {
  int n  = col0 + (lane & 15);
  int kb = (lane < 16) ? 0 : 8;
  const __bf16* p = BsT + n * ldb + kb;
  v16bf b;
#pragma unroll
  for (int e = 0; e < 8; ++e) { b[e] = p[e]; b[8 + e] = p[16 + e]; }
  return b;
}

static DEV v8f wmma_bf16(v16bf a, v16bf b, v8f c) {
  return __builtin_amdgcn_wmma_f32_16x16x32_bf16(false, a, false, b,
                                                 (short)0, c, false, false);
}

// copy 8 contiguous bf16 (16B) global -> LDS (fallback when no async DMA)
static DEV void copy8_bf16(const __bf16* __restrict__ g, __bf16* l) {
  uint4 q = *(const uint4*)g;
  const __bf16* qb = (const __bf16*)&q;
#pragma unroll
  for (int e = 0; e < 8; ++e) l[e] = qb[e];
}

// ---------------- prep kernels ----------------
__global__ __launch_bounds__(256) void k_zero32(unsigned int* p, int n) {
  int i = blockIdx.x * 256 + threadIdx.x;
  if (i < n) p[i] = 0u;
}

__global__ __launch_bounds__(256) void k_cvt_bf16(const float* __restrict__ s,
                                                  __bf16* __restrict__ d, int n) {
  int i = blockIdx.x * 256 + threadIdx.x;
  if (i < n) d[i] = f2bf(s[i]);
}

// Woff[t][m][c], m<144 -> off_w[m][c][t]; 144<=m<216 -> mask_w[m-144][c][t]; else 0
__global__ __launch_bounds__(256) void k_build_woff(const float* __restrict__ off_w,
                                                    const float* __restrict__ mask_w,
                                                    __bf16* __restrict__ Woff) {
  int i = blockIdx.x * 256 + threadIdx.x;      // < 9*224*256
  int t   = i / (MPAD * 256);
  int rem = i - t * (MPAD * 256);
  int m = rem >> 8, c = rem & 255;
  float v = 0.f;
  if (m < 144)      v = off_w[((size_t)m * 256 + c) * 9 + t];
  else if (m < 216) v = mask_w[((size_t)(m - 144) * 256 + c) * 9 + t];
  Woff[i] = f2bf(v);
}

// Wout[g][t][o][c] = out_w[(g*32+o)][c][t]
__global__ __launch_bounds__(256) void k_build_wout(const float* __restrict__ out_w,
                                                    __bf16* __restrict__ Wout) {
  int i = blockIdx.x * 256 + threadIdx.x;      // < 8*9*32*32
  int g    = i / 9216;
  int rem  = i - g * 9216;
  int t    = rem / 1024;
  int rem2 = rem - t * 1024;
  int o = rem2 >> 5, c = rem2 & 31;
  Wout[i] = f2bf(out_w[(((size_t)g * 32 + o) * 32 + c) * 9 + t]);
}

__global__ __launch_bounds__(256) void k_bias224(const float* __restrict__ off_b,
                                                 const float* __restrict__ mask_b,
                                                 float* __restrict__ bias) {
  int i = threadIdx.x;
  if (i < MPAD) {
    float v = 0.f;
    if (i < 144)      v = off_b[i];
    else if (i < 216) v = mask_b[i - 144];
    bias[i] = v;
  }
}

// ---------------- stage 1: in_proj GEMM  inp[b] = W(256x256) x X[b](256x4096) ----------------
__global__ __launch_bounds__(256) void k_inproj(const float* __restrict__ x,
                                                const __bf16* __restrict__ Wbf,
                                                const float* __restrict__ bvec,
                                                float* __restrict__ inp) {
  __shared__ alignas(16) __bf16 As[64][48];    // 48-el rows: every 8-el chunk 16B aligned
  __shared__ alignas(16) __bf16 Bs[64][36];
  int tid = threadIdx.x, lane = tid & 31, wave = tid >> 5;
  int wm = wave >> 1, wn = wave & 1;           // 4x2 wave grid, wave tile 16x32
  int b = blockIdx.z, mb = blockIdx.y * 64, nb = blockIdx.x * 64;
  v8f acc0 = {}, acc1 = {};
  for (int k0 = 0; k0 < 256; k0 += 32) {
    __syncthreads();
    {
      int r = tid >> 2, kk8 = (tid & 3) * 8;   // 64 rows x 4 chunks of 8 bf16
      const __bf16* gsrc = Wbf + (size_t)(mb + r) * 256 + k0 + kk8;
#if HAVE_ASYNC_LDS
      __builtin_amdgcn_global_load_async_to_lds_b128(AS1V(gsrc), AS3V(&As[r][kk8]), 0, 0);
#else
      copy8_bf16(gsrc, &As[r][kk8]);
#endif
    }
    // B: fp32 -> bf16, transposed store  (64 n x 32 k)
#pragma unroll
    for (int i = 0; i < 2; ++i) {
      int id = tid + i * 256;
      int k = id >> 4, n4 = (id & 15) * 4;
      const float4 v = *(const float4*)(x + ((size_t)b * CC + k0 + k) * HW + nb + n4);
      Bs[n4 + 0][k] = f2bf(v.x);
      Bs[n4 + 1][k] = f2bf(v.y);
      Bs[n4 + 2][k] = f2bf(v.z);
      Bs[n4 + 3][k] = f2bf(v.w);
    }
    if (k0 + 32 < 256)
      __builtin_prefetch(x + ((size_t)b * CC + k0 + 32 + (tid >> 4)) * HW + nb + (tid & 15) * 4, 0, 1);
    async_wait_all();
    __syncthreads();
    v16bf a  = load_a_frag(&As[0][0], 48, wm * 16, lane);
    v16bf b0 = load_b_frag(&Bs[0][0], 36, wn * 32, lane);
    v16bf b1 = load_b_frag(&Bs[0][0], 36, wn * 32 + 16, lane);
    acc0 = wmma_bf16(a, b0, acc0);
    acc1 = wmma_bf16(a, b1, acc1);
  }
  int mhalf = (lane < 16) ? 0 : 8;
  int ncol  = lane & 15;
#pragma unroll
  for (int j = 0; j < 8; ++j) {
    int m = mb + wm * 16 + j + mhalf;
    float bias = bvec[m];
    size_t base = ((size_t)b * CC + m) * HW + nb + wn * 32;
    inp[base + ncol]      = acc0[j] + bias;
    inp[base + 16 + ncol] = acc1[j] + bias;
  }
}

// ---------------- stage 2: depthwise 3x3 + partial LN stats ----------------
static DEV float dwconv_at(const float* __restrict__ xb, const float* __restrict__ dww,
                           float bias, int c, int h, int w) {
  float s = bias;
#pragma unroll
  for (int ky = 0; ky < 3; ++ky) {
    int yy = h + ky - 1;
#pragma unroll
    for (int kx = 0; kx < 3; ++kx) {
      int xx = w + kx - 1;
      if (yy >= 0 && yy < HH && xx >= 0 && xx < WW)
        s += xb[yy * WW + xx] * dww[c * 9 + ky * 3 + kx];
    }
  }
  return s;
}

__global__ __launch_bounds__(256) void k_dwstats(const float* __restrict__ x,
                                                 const float* __restrict__ dww,
                                                 const float* __restrict__ dwb,
                                                 float* __restrict__ partial) {
  int tid = threadIdx.x, b = blockIdx.y;
  int e = blockIdx.x * 256 + tid;              // < C*HW
  int c = e >> 12, hw = e & 4095, h = hw >> 6, w = hw & 63;
  const float* xb = x + ((size_t)b * CC + c) * HW;
  float s = dwconv_at(xb, dww, dwb[c], c, h, w);
  __shared__ float r1[256];
  __shared__ float r2[256];
  r1[tid] = s; r2[tid] = s * s;
  __syncthreads();
  for (int st = 128; st > 0; st >>= 1) {
    if (tid < st) { r1[tid] += r1[tid + st]; r2[tid] += r2[tid + st]; }
    __syncthreads();
  }
  if (tid == 0) {
    size_t pi = ((size_t)b * 4096 + blockIdx.x) * 2;
    partial[pi] = r1[0]; partial[pi + 1] = r2[0];
  }
}

__global__ __launch_bounds__(256) void k_reduce_stats(const float* __restrict__ partial,
                                                      float* __restrict__ sums) {
  int b = blockIdx.x, tid = threadIdx.x;
  float s = 0.f, s2 = 0.f;
  for (int i = tid; i < 4096; i += 256) {
    size_t pi = ((size_t)b * 4096 + i) * 2;
    s += partial[pi]; s2 += partial[pi + 1];
  }
  __shared__ float r1[256];
  __shared__ float r2[256];
  r1[tid] = s; r2[tid] = s2;
  __syncthreads();
  for (int st = 128; st > 0; st >>= 1) {
    if (tid < st) { r1[tid] += r1[tid + st]; r2[tid] += r2[tid + st]; }
    __syncthreads();
  }
  if (tid == 0) { sums[b * 2] = r1[0]; sums[b * 2 + 1] = r2[0]; }
}

// ---------------- stage 3: LN + exact GELU -> padded bf16 ----------------
__global__ __launch_bounds__(256) void k_lngelu(const float* __restrict__ x,
                                                const float* __restrict__ dww,
                                                const float* __restrict__ dwb,
                                                const float* __restrict__ sums,
                                                __bf16* __restrict__ xpad) {
  int tid = threadIdx.x, b = blockIdx.y;
  int e = blockIdx.x * 256 + tid;
  int c = e >> 12, hw = e & 4095, h = hw >> 6, w = hw & 63;
  const float* xb = x + ((size_t)b * CC + c) * HW;
  float s = dwconv_at(xb, dww, dwb[c], c, h, w);
  const float invN = 1.0f / (float)(CC * HW);
  float mu  = sums[b * 2] * invN;
  float var = sums[b * 2 + 1] * invN - mu * mu;
  float tn  = (s - mu) * rsqrtf(var + 1e-5f);
  float ge  = 0.5f * tn * (1.0f + erff(tn * 0.70710678118654752f));
  xpad[(((size_t)b * CC + c) * WPAD + (h + 1)) * WPAD + (w + 1)] = f2bf(ge);
}

// ---------------- stage 4: offset+mask heads as 9 shifted 1x1 GEMMs ----------------
__global__ __launch_bounds__(256) void k_offmask(const __bf16* __restrict__ xpad,
                                                 const __bf16* __restrict__ Woff,
                                                 const float* __restrict__ bias224,
                                                 float* __restrict__ OM) {
  __shared__ alignas(16) __bf16 As[32][48];
  __shared__ alignas(16) __bf16 Bs[128][36];
  int tid = threadIdx.x, lane = tid & 31, wave = tid >> 5;
  int wm = wave >> 2, wn = wave & 3;           // 2x4 wave grid, block tile 32x128
  int b = blockIdx.z, mb = blockIdx.y * 32, nb = blockIdx.x * 128;
  v8f acc0 = {}, acc1 = {};
  for (int t = 0; t < 9; ++t) {
    int ty = t / 3, tx = t % 3;
    for (int k0 = 0; k0 < 256; k0 += 32) {
      __syncthreads();
      if (tid < 128) {                          // A panel: pure bf16 memcpy
        int r = tid >> 2, kk8 = (tid & 3) * 8;
        const __bf16* gsrc = Woff + ((size_t)t * MPAD + mb + r) * 256 + k0 + kk8;
#if HAVE_ASYNC_LDS
        __builtin_amdgcn_global_load_async_to_lds_b128(AS1V(gsrc), AS3V(&As[r][kk8]), 0, 0);
#else
        copy8_bf16(gsrc, &As[r][kk8]);
#endif
      }
#pragma unroll
      for (int i = 0; i < 16; ++i) {            // B panel: shifted, unaligned -> scalar
        int idx = tid + i * 256;
        int n = idx & 127, k = idx >> 7;
        int pix = nb + n;
        int h = pix >> 6, w = pix & 63;
        Bs[n][k] = xpad[(((size_t)b * CC + k0 + k) * WPAD + h + ty) * WPAD + w + tx];
      }
      if (k0 + 32 < 256) {
        int pix = nb + (tid & 127);
        int h = pix >> 6, w = pix & 63;
        __builtin_prefetch(xpad + (((size_t)b * CC + k0 + 32 + (tid >> 7)) * WPAD + h + ty) * WPAD + w + tx, 0, 1);
      }
      async_wait_all();
      __syncthreads();
      v16bf a  = load_a_frag(&As[0][0], 48, wm * 16, lane);
      v16bf b0 = load_b_frag(&Bs[0][0], 36, wn * 32, lane);
      v16bf b1 = load_b_frag(&Bs[0][0], 36, wn * 32 + 16, lane);
      acc0 = wmma_bf16(a, b0, acc0);
      acc1 = wmma_bf16(a, b1, acc1);
    }
  }
  int mhalf = (lane < 16) ? 0 : 8;
  int ncol  = lane & 15;
#pragma unroll
  for (int j = 0; j < 8; ++j) {
    int ch = mb + wm * 16 + j + mhalf;
    float bias = bias224[ch];
    size_t base = ((size_t)b * MPAD + ch) * HW + nb + wn * 32;
    OM[base + ncol]      = acc0[j] + bias;
    OM[base + 16 + ncol] = acc1[j] + bias;
  }
}

// ---------------- stage 5: softmax over consecutive 9 (faithful reshape) ----------------
__global__ __launch_bounds__(256) void k_softmax9(float* __restrict__ OM) {
  int b = blockIdx.y;
  int gi = blockIdx.x * 256 + threadIdx.x;     // < 72*4096/9 = 32768
  float* p = OM + ((size_t)b * MPAD + 144) * HW + (size_t)gi * 9;
  float m = p[0];
#pragma unroll
  for (int i = 1; i < 9; ++i) m = fmaxf(m, p[i]);
  float e[9]; float s = 0.f;
#pragma unroll
  for (int i = 0; i < 9; ++i) { e[i] = expf(p[i] - m); s += e[i]; }
  float inv = 1.0f / s;
#pragma unroll
  for (int i = 0; i < 9; ++i) p[i] = e[i] * inv;
}

// ---------------- stage 6: fused bilinear sampling + mask + grouped output WMMA GEMM ----------------
__global__ __launch_bounds__(128) void k_sample_out(const float* __restrict__ inp,
                                                    const float* __restrict__ OM,
                                                    const __bf16* __restrict__ Wout,
                                                    const float* __restrict__ outb,
                                                    float* __restrict__ out) {
  __shared__ alignas(16) __bf16 As9[9][32][34];  // out_w per tap, 32(o) x 32(c)
  __shared__ alignas(16) __bf16 Vs[64][36];      // v tile transposed: [pixel][c]
  __shared__ int   sIdx[4][64];
  __shared__ float sWgt[4][64];
  int tid = threadIdx.x, lane = tid & 31, wave = tid >> 5;
  int h = blockIdx.x, g = blockIdx.y, b = blockIdx.z;
  // stage all 9 tap weights
#pragma unroll 4
  for (int i = 0; i < 72; ++i) {
    int idx = tid + i * 128;                   // < 9216
    int t = idx >> 10, rem = idx & 1023;
    int r = rem >> 5, c = rem & 31;
    As9[t][r][c] = Wout[(((size_t)g * 9 + t) * 32 + r) * 32 + c];
  }
  v8f acc0 = {}, acc1 = {};
  int m0  = (wave >> 1) * 16;                  // 2x2 wave grid over 32(o) x 64(pix)
  int nb0 = (wave & 1) * 32;
  const float* OMb = OM + (size_t)b * MPAD * HW;
  size_t hw_base = (size_t)h * WW;
  for (int t = 0; t < 9; ++t) {
    int ki = t / 3, kj = t % 3;
    __syncthreads();
    if (tid < 64) {
      int p = tid;
      size_t hw = hw_base + p;
      int cho = (g * 9 + t) * 2;
      float dy = OMb[(size_t)cho * HW + hw];
      float dx = OMb[(size_t)(cho + 1) * HW + hw];
      float mv = OMb[(size_t)(144 + g * 9 + t) * HW + hw];
      float py = dy + (float)(ki + h - 1);
      float px = dx + (float)(kj + p - 1);
      float fy0 = floorf(py), fx0 = floorf(px);
      float wy = py - fy0, wx = px - fx0;
      int iy0 = (int)fy0, ix0 = (int)fx0;
      int iy1 = iy0 + 1,  ix1 = ix0 + 1;
      bool vy0 = (iy0 >= 0) && (iy0 < HH), vy1 = (iy1 >= 0) && (iy1 < HH);
      bool vx0 = (ix0 >= 0) && (ix0 < WW), vx1 = (ix1 >= 0) && (ix1 < WW);
      int cy0 = min(max(iy0, 0), HH - 1), cy1 = min(max(iy1, 0), HH - 1);
      int cx0 = min(max(ix0, 0), WW - 1), cx1 = min(max(ix1, 0), WW - 1);
      sIdx[0][p] = cy0 * WW + cx0; sWgt[0][p] = (vy0 && vx0) ? (1.f - wy) * (1.f - wx) * mv : 0.f;
      sIdx[1][p] = cy0 * WW + cx1; sWgt[1][p] = (vy0 && vx1) ? (1.f - wy) * wx * mv : 0.f;
      sIdx[2][p] = cy1 * WW + cx0; sWgt[2][p] = (vy1 && vx0) ? wy * (1.f - wx) * mv : 0.f;
      sIdx[3][p] = cy1 * WW + cx1; sWgt[3][p] = (vy1 && vx1) ? wy * wx * mv : 0.f;
    }
    __syncthreads();
    {
      int p  = tid & 63;
      int c0 = tid >> 6;                       // 0 or 1
      int i0 = sIdx[0][p], i1 = sIdx[1][p], i2 = sIdx[2][p], i3 = sIdx[3][p];
      float w0 = sWgt[0][p], w1 = sWgt[1][p], w2 = sWgt[2][p], w3 = sWgt[3][p];
#pragma unroll
      for (int i = 0; i < 16; ++i) {
        int c = c0 + i * 2;
        const float* src = inp + (((size_t)b * GG + g) * CG + c) * HW;
        float v = w0 * src[i0] + w1 * src[i1] + w2 * src[i2] + w3 * src[i3];
        Vs[p][c] = f2bf(v);
      }
    }
    __syncthreads();
    v16bf a  = load_a_frag(&As9[t][0][0], 34, m0, lane);
    v16bf b0 = load_b_frag(&Vs[0][0], 36, nb0, lane);
    v16bf b1 = load_b_frag(&Vs[0][0], 36, nb0 + 16, lane);
    acc0 = wmma_bf16(a, b0, acc0);
    acc1 = wmma_bf16(a, b1, acc1);
  }
  int mhalf = (lane < 16) ? 0 : 8;
  int ncol  = lane & 15;
#pragma unroll
  for (int j = 0; j < 8; ++j) {
    int ch = g * CG + m0 + j + mhalf;
    float bias = outb[ch];
    size_t base = ((size_t)b * CC + ch) * HW + hw_base + nb0;
    out[base + ncol]      = acc0[j] + bias;
    out[base + 16 + ncol] = acc1[j] + bias;
  }
}

// ---------------- host launcher ----------------
extern "C" void kernel_launch(void* const* d_in, const int* in_sizes, int n_in,
                              void* d_out, int out_size, void* d_ws, size_t ws_size,
                              hipStream_t stream) {
  (void)in_sizes; (void)n_in; (void)out_size; (void)ws_size;
  const float* x         = (const float*)d_in[0];
  const float* in_proj_w = (const float*)d_in[1];
  const float* in_proj_b = (const float*)d_in[2];
  const float* dw_w      = (const float*)d_in[3];
  const float* dw_b      = (const float*)d_in[4];
  const float* off_w     = (const float*)d_in[5];
  const float* off_b     = (const float*)d_in[6];
  const float* mask_w    = (const float*)d_in[7];
  const float* mask_b    = (const float*)d_in[8];
  const float* out_w     = (const float*)d_in[9];
  const float* out_b     = (const float*)d_in[10];
  float* out = (float*)d_out;

  char* ws = (char*)d_ws;
  size_t off = 0;
  auto carve = [&](size_t bytes) -> char* {
    char* p = ws + off;
    off += (bytes + 255) & ~(size_t)255;
    return p;
  };
  float*  inp     = (float*) carve((size_t)BB * CC * HW * 4);          // 16.78 MB
  __bf16* xpad    = (__bf16*)carve((size_t)BB * CC * WPAD * WPAD * 2); // 8.92 MB
  float*  OM      = (float*) carve((size_t)BB * MPAD * HW * 4);        // 14.68 MB
  __bf16* Wbf     = (__bf16*)carve((size_t)CC * CC * 2);
  __bf16* Woff    = (__bf16*)carve((size_t)9 * MPAD * CC * 2);
  __bf16* Wout    = (__bf16*)carve((size_t)GG * 9 * CG * CG * 2);
  float*  bias224 = (float*) carve((size_t)MPAD * 4);
  float*  sums    = (float*) carve(256);
  float*  partial = (float*) carve((size_t)BB * 4096 * 2 * 4);

  // --- prep (independent) ---
  {
    int ndw = (int)((size_t)BB * CC * WPAD * WPAD * 2 / 4);            // xpad as dwords
    k_zero32<<<(ndw + 255) / 256, 256, 0, stream>>>((unsigned int*)xpad, ndw);
  }
  k_cvt_bf16 <<<(CC * CC + 255) / 256, 256, 0, stream>>>(in_proj_w, Wbf, CC * CC);
  k_build_woff<<<(9 * MPAD * CC + 255) / 256, 256, 0, stream>>>(off_w, mask_w, Woff);
  k_build_wout<<<(GG * 9 * CG * CG + 255) / 256, 256, 0, stream>>>(out_w, Wout);
  k_bias224  <<<1, 256, 0, stream>>>(off_b, mask_b, bias224);

  // --- stage 1: in_proj GEMM ---
  k_inproj<<<dim3(HW / 64, CC / 64, BB), 256, 0, stream>>>(x, Wbf, in_proj_b, inp);

  // --- stages 2-3: depthwise conv + LayerNorm + GELU ---
  k_dwstats     <<<dim3(4096, BB), 256, 0, stream>>>(x, dw_w, dw_b, partial);
  k_reduce_stats<<<BB, 256, 0, stream>>>(partial, sums);
  k_lngelu      <<<dim3(4096, BB), 256, 0, stream>>>(x, dw_w, dw_b, sums, xpad);

  // --- stage 4: offset + mask heads (WMMA GEMM, 9 shifted taps) ---
  k_offmask<<<dim3(HW / 128, MPAD / 32, BB), 256, 0, stream>>>(xpad, Woff, bias224, OM);

  // --- stage 5: softmax over consecutive 9 (in place on mask rows) ---
  k_softmax9<<<dim3(32768 / 256, BB), 256, 0, stream>>>(OM);

  // --- stage 6: fused sampling + masked accumulation + output GEMM ---
  k_sample_out<<<dim3(HH, GG, BB), 128, 0, stream>>>(inp, OM, Wout, out_b, out);
}
